// UniBidirectional_Spatial_Attention_51625506898203
// MI455X (gfx1250) — matrix operand order repaired and checked by reading, not compiled, not run on hardware
//
#include <hip/hip_runtime.h>
#include <hip/hip_bf16.h>

typedef __attribute__((ext_vector_type(16))) _Float16 v16h;
typedef __attribute__((ext_vector_type(8)))  _Float16 v8h;
typedef __attribute__((ext_vector_type(8)))  float    v8f;
typedef __attribute__((ext_vector_type(4)))  float    v4f;

#define N_TOK 4096
#define CH    64

// Build a 16-half WMMA fragment from two 16-byte runs.
__device__ __forceinline__ v16h frag_from2(const _Float16* p0, const _Float16* p1) {
  v8h a = *(const v8h*)p0;
  v8h b = *(const v8h*)p1;
  return __builtin_shufflevector(a, b, 0,1,2,3,4,5,6,7,8,9,10,11,12,13,14,15);
}

// ---------------------------------------------------------------------------
// 1) Pointwise channel projections: x(b,c,n) * W(o,c) -> f16
//    which: 0=q_tt 1=k_tt 2=v_tt(T) 3=q_st 4=k_st 5=v_st(T)
// ---------------------------------------------------------------------------
__global__ __launch_bounds__(256) void proj_kernel(
    const float* __restrict__ src, const float* __restrict__ tgt,
    const float* __restrict__ Wqtt, const float* __restrict__ Wqst,
    const float* __restrict__ Wktt, const float* __restrict__ Wkst,
    const float* __restrict__ Wvtt, const float* __restrict__ Wvst,
    _Float16* __restrict__ qtt, _Float16* __restrict__ ktt, _Float16* __restrict__ vtt,
    _Float16* __restrict__ qst, _Float16* __restrict__ kst, _Float16* __restrict__ vst)
{
  const int which = blockIdx.y;
  const unsigned idx = blockIdx.x * 256u + threadIdx.x;   // over B*N*CH = 524288
  const int o = idx & 63;
  const int n = (idx >> 6) & 4095;
  const int b = idx >> 18;

  const float* x; const float* W; _Float16* dst; bool tr;
  switch (which) {
    case 0:  x = tgt; W = Wqtt; dst = qtt; tr = false; break;
    case 1:  x = tgt; W = Wktt; dst = ktt; tr = false; break;
    case 2:  x = tgt; W = Wvtt; dst = vtt; tr = true;  break;
    case 3:  x = tgt; W = Wqst; dst = qst; tr = false; break;
    case 4:  x = src; W = Wkst; dst = kst; tr = false; break;
    default: x = src; W = Wvst; dst = vst; tr = true;  break;
  }
  const float* xp = x + (size_t)b * CH * N_TOK + n;
  const float* wp = W + o * CH;
  float acc = 0.0f;
#pragma unroll 8
  for (int c = 0; c < CH; ++c) acc += xp[(size_t)c * N_TOK] * wp[c];
  const _Float16 h = (_Float16)acc;
  if (tr) dst[((size_t)b * CH + o) * N_TOK + n] = h;   // V stored transposed (b,c,n)
  else    dst[((size_t)b * N_TOK + n) * CH + o] = h;   // Q/K stored (b,n,c)
}

// ---------------------------------------------------------------------------
// 2) Flash attention: one wave per 16-query tile; key blocks of 32.
//    Q,K: (b,N,64) f16 ; Vt: (b,64,N) f16 ; att out: (b,N,64) f32
// ---------------------------------------------------------------------------
__global__ __launch_bounds__(256) void flash_attn_kernel(
    const _Float16* __restrict__ Q, const _Float16* __restrict__ K,
    const _Float16* __restrict__ Vt, float* __restrict__ att)
{
  __shared__ float sS[8][16 * 32];   // per-wave score tile (16 rows x 32 keys)
  __shared__ float sR[8][16];        // per-wave row-broadcast scratch

  const int tid  = threadIdx.x;
  const int wave = tid >> 5;
  const int lane = tid & 31;
  const int half = lane >> 4;        // 0 or 1 (half-wave)
  const int lm   = lane & 15;
  const int koff = half << 3;        // 0 or 8; unused (A operands also use boff)
  const int boff = half << 4;        // B-operand K offset (contiguous 16)

  const int tile = blockIdx.x * 8 + wave;   // 512 tiles = B * N/16
  const int b    = tile >> 8;
  const int q0   = (tile & 255) << 4;

  const _Float16* Qb = Q  + (size_t)b * N_TOK * CH;
  const _Float16* Kb = K  + (size_t)b * N_TOK * CH;
  const _Float16* Vb = Vt + (size_t)b * CH * N_TOK;
  float* sSc = sS[wave];
  float* sRw = sR[wave];

  // Q A-fragments (rows q0..q0+15, c = 0..31 and 32..63), loaded once.
  const _Float16* qp = Qb + (size_t)(q0 + lm) * CH;
  const v16h qa0 = frag_from2(qp + boff,      qp + boff + 8);
  const v16h qa1 = frag_from2(qp + 32 + boff, qp + 40 + boff);

  // V row pointers (channel tiles), fixed per wave.
  const _Float16* vr0 = Vb + (size_t)(lm)      * N_TOK + boff;
  const _Float16* vr1 = Vb + (size_t)(16 + lm) * N_TOK + boff;
  const _Float16* vr2 = Vb + (size_t)(32 + lm) * N_TOK + boff;
  const _Float16* vr3 = Vb + (size_t)(48 + lm) * N_TOK + boff;

  v8f o0 = {}, o1 = {}, o2 = {}, o3 = {};
  const v8f z = {};
  float m_run = -__builtin_inff();
  float l_run = 0.0f;
  const float scale = 0.125f;        // 1/sqrt(64)

  for (int kb0 = 0; kb0 < N_TOK; kb0 += 32) {
    // ---- issue ALL fragment loads for this block up front (one clause) ----
    const _Float16* kp0 = Kb + (size_t)(kb0 + lm) * CH;       // keys kb0..+15
    const _Float16* kp1 = Kb + (size_t)(kb0 + 16 + lm) * CH;  // keys kb0+16..+31
    const v16h kf00 = frag_from2(kp0 + boff,      kp0 + boff + 8);
    const v16h kf01 = frag_from2(kp0 + 32 + boff, kp0 + 40 + boff);
    const v16h kf10 = frag_from2(kp1 + boff,      kp1 + boff + 8);
    const v16h kf11 = frag_from2(kp1 + 32 + boff, kp1 + 40 + boff);
    // V fragments prefetched here: their latency is hidden behind the
    // LDS bounce + softmax VALU work below.
    const v16h vf0 = frag_from2(vr0 + kb0, vr0 + kb0 + 8);
    const v16h vf1 = frag_from2(vr1 + kb0, vr1 + kb0 + 8);
    const v16h vf2 = frag_from2(vr2 + kb0, vr2 + kb0 + 8);
    const v16h vf3 = frag_from2(vr3 + kb0, vr3 + kb0 + 8);

    // ---- S = Q * K^T for this 32-key block (two 16-key column tiles) ----
    v8f sc0 = __builtin_amdgcn_wmma_f32_16x16x32_f16(false, qa0, false, kf00, (short)0, z,   false, false);
    sc0     = __builtin_amdgcn_wmma_f32_16x16x32_f16(false, qa1, false, kf01, (short)0, sc0, false, false);
    v8f sc1 = __builtin_amdgcn_wmma_f32_16x16x32_f16(false, qa0, false, kf10, (short)0, z,   false, false);
    sc1     = __builtin_amdgcn_wmma_f32_16x16x32_f16(false, qa1, false, kf11, (short)0, sc1, false, false);

    // ---- C-layout -> LDS (m,n) ----
#pragma unroll
    for (int r = 0; r < 8; ++r) {
      const int m = r + (half << 3);
      sSc[m * 32 + lm]      = sc0[r];
      sSc[m * 32 + 16 + lm] = sc1[r];
    }

    // ---- read back in A-operand order: lane = row lm, 16 of 32 keys ----
    float s[16];
    {
      const v4f x0 = *(const v4f*)&sSc[lm * 32 + boff];
      const v4f x1 = *(const v4f*)&sSc[lm * 32 + boff + 4];
      const v4f y0 = *(const v4f*)&sSc[lm * 32 + 8 + boff];
      const v4f y1 = *(const v4f*)&sSc[lm * 32 + 12 + boff];
#pragma unroll
      for (int i = 0; i < 4; ++i) {
        s[i]      = x0[i] * scale;  s[4 + i]  = x1[i] * scale;
        s[8 + i]  = y0[i] * scale;  s[12 + i] = y1[i] * scale;
      }
    }

    // ---- online softmax (row stats shared across half-waves) ----
    float bm = s[0];
#pragma unroll
    for (int i = 1; i < 16; ++i) bm = fmaxf(bm, s[i]);
    bm = fmaxf(bm, __shfl_xor(bm, 16, 32));
    const float m_new = fmaxf(m_run, bm);
    const float alpha = __expf(m_run - m_new);

    v16h pa;
    float ls = 0.0f;
#pragma unroll
    for (int i = 0; i < 16; ++i) {
      const float p = __expf(s[i] - m_new);
      ls += p;
      pa[i] = (_Float16)p;
    }
    ls += __shfl_xor(ls, 16, 32);
    l_run = l_run * alpha + ls;
    m_run = m_new;

    // ---- rescale O accumulators (per-row alpha broadcast via LDS) ----
    if (half == 0) sRw[lm] = alpha;
    {
      const v4f a0 = *(const v4f*)&sRw[half << 3];
      const v4f a1 = *(const v4f*)&sRw[(half << 3) + 4];
#pragma unroll
      for (int r = 0; r < 4; ++r) {
        o0[r] *= a0[r]; o0[4 + r] *= a1[r];
        o1[r] *= a0[r]; o1[4 + r] *= a1[r];
        o2[r] *= a0[r]; o2[4 + r] *= a1[r];
        o3[r] *= a0[r]; o3[4 + r] *= a1[r];
      }
    }

    // ---- O += P * V (four 16-channel column tiles) ----
    o0 = __builtin_amdgcn_wmma_f32_16x16x32_f16(false, pa, false, vf0, (short)0, o0, false, false);
    o1 = __builtin_amdgcn_wmma_f32_16x16x32_f16(false, pa, false, vf1, (short)0, o1, false, false);
    o2 = __builtin_amdgcn_wmma_f32_16x16x32_f16(false, pa, false, vf2, (short)0, o2, false, false);
    o3 = __builtin_amdgcn_wmma_f32_16x16x32_f16(false, pa, false, vf3, (short)0, o3, false, false);
  }

  // ---- normalize by row sums and write out ----
  if (half == 0) sRw[lm] = 1.0f / l_run;
  const v4f l0 = *(const v4f*)&sRw[half << 3];
  const v4f l1 = *(const v4f*)&sRw[(half << 3) + 4];
  float* ap = att + ((size_t)b * N_TOK + q0) * CH;
#pragma unroll
  for (int r = 0; r < 8; ++r) {
    const int m = r + (half << 3);
    const float lv = (r < 4) ? l0[r] : l1[r - 4];
    ap[(size_t)m * CH + lm]      = o0[r] * lv;
    ap[(size_t)m * CH + 16 + lm] = o1[r] * lv;
    ap[(size_t)m * CH + 32 + lm] = o2[r] * lv;
    ap[(size_t)m * CH + 48 + lm] = o3[r] * lv;
  }
}

// ---------------------------------------------------------------------------
// 3) out[b,o,n] = sum_c Wout[o,c] * 0.5*(att1+att2)[b,n,c]
// ---------------------------------------------------------------------------
__global__ __launch_bounds__(256) void out_proj_kernel(
    const float* __restrict__ att1, const float* __restrict__ att2,
    const float* __restrict__ Wout, float* __restrict__ out)
{
  const unsigned idx = blockIdx.x * 256u + threadIdx.x;  // B*64*N = 524288
  const int n = idx & 4095;
  const int o = (idx >> 12) & 63;
  const int b = idx >> 18;
  const float* a1 = att1 + ((size_t)b * N_TOK + n) * CH;
  const float* a2 = att2 + ((size_t)b * N_TOK + n) * CH;
  const float* wp = Wout + o * CH;
  float acc = 0.0f;
#pragma unroll 8
  for (int c = 0; c < CH; ++c) acc += wp[c] * (a1[c] + a2[c]);
  out[idx] = 0.5f * acc;   // idx == (b*64 + o)*N + n
}

extern "C" void kernel_launch(void* const* d_in, const int* in_sizes, int n_in,
                              void* d_out, int out_size, void* d_ws, size_t ws_size,
                              hipStream_t stream) {
  (void)in_sizes; (void)n_in; (void)out_size; (void)ws_size;
  const float* src  = (const float*)d_in[0];
  const float* tgt  = (const float*)d_in[1];
  const float* Wqtt = (const float*)d_in[2];
  const float* Wqst = (const float*)d_in[3];
  const float* Wktt = (const float*)d_in[4];
  const float* Wkst = (const float*)d_in[5];
  const float* Wvtt = (const float*)d_in[6];
  const float* Wvst = (const float*)d_in[7];
  const float* Wout = (const float*)d_in[8];
  float* out = (float*)d_out;

  char* ws = (char*)d_ws;
  const size_t S16 = (size_t)2 * N_TOK * CH * sizeof(_Float16);   // 1 MiB each
  _Float16* qtt = (_Float16*)(ws + 0 * S16);
  _Float16* ktt = (_Float16*)(ws + 1 * S16);
  _Float16* vtt = (_Float16*)(ws + 2 * S16);
  _Float16* qst = (_Float16*)(ws + 3 * S16);
  _Float16* kst = (_Float16*)(ws + 4 * S16);
  _Float16* vst = (_Float16*)(ws + 5 * S16);
  const size_t S32 = (size_t)2 * N_TOK * CH * sizeof(float);      // 2 MiB each
  float* att1 = (float*)(ws + 6 * S16);
  float* att2 = (float*)(ws + 6 * S16 + S32);

  proj_kernel<<<dim3(2048, 6), 256, 0, stream>>>(src, tgt, Wqtt, Wqst, Wktt, Wkst,
                                                 Wvtt, Wvst, qtt, ktt, vtt, qst, kst, vst);
  flash_attn_kernel<<<64, 256, 0, stream>>>(qtt, ktt, vtt, att1);
  flash_attn_kernel<<<64, 256, 0, stream>>>(qst, kst, vst, att2);
  out_proj_kernel<<<2048, 256, 0, stream>>>(att1, att2, Wout, out);
}
